// Model_90967407329786
// MI455X (gfx1250) — compile-verified, matrix operand order checked by reading, not run on hardware
//
#include <hip/hip_runtime.h>

typedef __attribute__((ext_vector_type(16))) _Float16 v16h;
typedef __attribute__((ext_vector_type(8)))  _Float16 v8h;
typedef __attribute__((ext_vector_type(8)))  float    v8f;
typedef __attribute__((ext_vector_type(4)))  int      v4i_t;

#define TPB 256
#define GRID1(tot) dim3((unsigned)((((long)(tot)) + TPB - 1) / TPB))

#if defined(__has_builtin)
#  if __has_builtin(__builtin_amdgcn_global_load_async_to_lds_b128) && \
      __has_builtin(__builtin_amdgcn_s_wait_asynccnt)
#    define HAVE_ASYNC_LDS 1
#  endif
#endif
#ifndef HAVE_ASYNC_LDS
#  define HAVE_ASYNC_LDS 0
#endif

#define AS1 __attribute__((address_space(1)))
#define AS3 __attribute__((address_space(3)))

static inline long rnd(long x, long m) { return (x + m - 1) / m * m; }

// K-major WMMA fragment: lane group g needs K[g*8..g*8+7] and K[16+g*8..+7]
// of one row, sequential order -> two contiguous ds_load_b128 + shuffle.
__device__ __forceinline__ v16h frag16(const _Float16* base, int row, int g)
{
    const int off = row * 32 + g * 8;
    v8h lo = *reinterpret_cast<const v8h*>(base + off);
    v8h hi = *reinterpret_cast<const v8h*>(base + off + 16);
    return __builtin_shufflevector(lo, hi, 0,1,2,3,4,5,6,7,8,9,10,11,12,13,14,15);
}

// ---------------------------------------------------------------------------
// GEMM: C[M,N] = A[M,K] * B[K,N].
// A f16 row-major [M,K] (lda); B supplied TRANSPOSED: BT f16 [N,K] (ldbt) so
// both operands are K-major. 256 threads = 8 waves (2x4); 64x128 C macro-tile;
// each wave computes 32x32 = four v_wmma_f32_16x16x32_f16 accumulators per
// K-step from 2 A + 2 B fragments (each fragment reused twice).
// Global->LDS staging is double-buffered via GLOBAL_LOAD_ASYNC_TO_LDS_B128:
// each iteration issues the NEXT tile's DMA, then waits only for the
// PREVIOUS fill (ASYNCcnt completes in order), overlapping DMA with WMMA.
// ---------------------------------------------------------------------------
__global__ void __launch_bounds__(256)
gemm_wmma_f16(const _Float16* __restrict__ A, const _Float16* __restrict__ BT,
              float* __restrict__ C, int M, int N, int K,
              int lda, int ldbt, long ldc)
{
    // per buffer: A tile 64x32 halves (4KB) + B tile 128x32 halves (8KB)
    const int BUF = 64 * 32 + 128 * 32;
    __shared__ __align__(16) _Float16 smem[2 * (64 * 32 + 128 * 32)];

    const int tid  = threadIdx.x;
    const int lane = tid & 31;
    const int wave = tid >> 5;
    const int wm   = (wave >> 2) * 32;   // wave row offset (2 wave-rows)
    const int wn   = (wave & 3) * 32;    // wave col offset (4 wave-cols)
    const int g    = lane >> 4;          // lane group (K-half select)
    const int lr   = lane & 15;

    const long m0 = (long)blockIdx.y * 64;
    const long n0 = (long)blockIdx.x * 128;

    v8f acc00 = {}, acc01 = {}, acc10 = {}, acc11 = {};

    // staging assignments: A = 1 b128/thread, B = 2 b128/thread
    const int ar  = tid >> 2,      ac  = (tid & 3) * 8;
    const int bc0 = tid * 2,       bc1 = tid * 2 + 1;
    const int br0 = bc0 >> 2,      bo0 = (bc0 & 3) * 8;
    const int br1 = bc1 >> 2,      bo1 = (bc1 & 3) * 8;

    const _Float16* gA  = A  + (m0 + ar)  * lda  + ac;
    const _Float16* gB0 = BT + (n0 + br0) * ldbt + bo0;
    const _Float16* gB1 = BT + (n0 + br1) * ldbt + bo1;

    _Float16* sA  = smem + ar * 32 + ac;
    _Float16* sB0 = smem + 64 * 32 + br0 * 32 + bo0;
    _Float16* sB1 = smem + 64 * 32 + br1 * 32 + bo1;

    auto fill = [&](int buf, int k0) {
#if HAVE_ASYNC_LDS
        __builtin_amdgcn_global_load_async_to_lds_b128(
            (AS1 v4i_t*)(gA  + k0), (AS3 v4i_t*)(sA  + buf * BUF), 0, 0);
        __builtin_amdgcn_global_load_async_to_lds_b128(
            (AS1 v4i_t*)(gB0 + k0), (AS3 v4i_t*)(sB0 + buf * BUF), 0, 0);
        __builtin_amdgcn_global_load_async_to_lds_b128(
            (AS1 v4i_t*)(gB1 + k0), (AS3 v4i_t*)(sB1 + buf * BUF), 0, 0);
#else
        *reinterpret_cast<float4*>(sA  + buf * BUF) = *reinterpret_cast<const float4*>(gA  + k0);
        *reinterpret_cast<float4*>(sB0 + buf * BUF) = *reinterpret_cast<const float4*>(gB0 + k0);
        *reinterpret_cast<float4*>(sB1 + buf * BUF) = *reinterpret_cast<const float4*>(gB1 + k0);
#endif
    };

    fill(0, 0);
    int buf = 0;
    for (int k0 = 0; k0 < K; k0 += 32, buf ^= 1) {
        if (k0 + 32 < K) {
            fill(buf ^ 1, k0 + 32);          // prefetch next tile (other buffer)
#if HAVE_ASYNC_LDS
            __builtin_amdgcn_s_wait_asynccnt(3);  // wait only for PREVIOUS fill
#endif
        } else {
#if HAVE_ASYNC_LDS
            __builtin_amdgcn_s_wait_asynccnt(0);
#endif
        }
        __syncthreads();

        const _Float16* Ab = smem + buf * BUF;
        const _Float16* Bb = Ab + 64 * 32;

        v16h a0 = frag16(Ab, wm + lr, g);
        v16h a1 = frag16(Ab, wm + 16 + lr, g);
        v16h b0 = frag16(Bb, wn + lr, g);
        v16h b1 = frag16(Bb, wn + 16 + lr, g);

        acc00 = __builtin_amdgcn_wmma_f32_16x16x32_f16(false, a0, false, b0,
                                                       (short)0, acc00, false, false);
        acc01 = __builtin_amdgcn_wmma_f32_16x16x32_f16(false, a0, false, b1,
                                                       (short)0, acc01, false, false);
        acc10 = __builtin_amdgcn_wmma_f32_16x16x32_f16(false, a1, false, b0,
                                                       (short)0, acc10, false, false);
        acc11 = __builtin_amdgcn_wmma_f32_16x16x32_f16(false, a1, false, b1,
                                                       (short)0, acc11, false, false);
        __syncthreads();   // all waves done reading buf before it is refilled
    }

#pragma unroll
    for (int r = 0; r < 8; ++r) {
        // C/D layout: VGPR r -> M = r + 8*laneGroup ; N = lane&15
        const long row0 = m0 + wm + r + g * 8;
        const long row1 = row0 + 16;
        const long col0 = n0 + wn + lr;
        const long col1 = col0 + 16;
        if (row0 < M) {
            if (col0 < N) C[row0 * ldc + col0] = acc00[r];
            if (col1 < N) C[row0 * ldc + col1] = acc01[r];
        }
        if (row1 < M) {
            if (col0 < N) C[row1 * ldc + col0] = acc10[r];
            if (col1 < N) C[row1 * ldc + col1] = acc11[r];
        }
    }
}

// ---------------------------------------------------------------------------
// Packers / im2col (f32 -> f16, zero-padded to WMMA-friendly shapes)
// ---------------------------------------------------------------------------
__global__ void pack_f16_k(_Float16* __restrict__ dst, const float* __restrict__ src,
                           long rows, long cols, long rstride, long cstride,
                           long Rp, long Cp)
{
    long i = blockIdx.x * (long)blockDim.x + threadIdx.x;
    if (i >= Rp * Cp) return;
    long r = i / Cp, c = i % Cp;
    _Float16 v = (_Float16)0.f;
    if (r < rows && c < cols) v = (_Float16)src[r * rstride + c * cstride];
    dst[i] = v;
}

// ConvTranspose weights (Cin,Cout,KH,KW) -> packed [Mp x Kp] with flip+transpose
__global__ void pack_wT_k(_Float16* __restrict__ dst, const float* __restrict__ w,
                          int Cin, int Cout, int KH, int KW, long Kp, long Mp)
{
    long i = blockIdx.x * (long)blockDim.x + threadIdx.x;
    if (i >= Mp * Kp) return;
    long m = i / Kp, k = i % Kp;
    _Float16 v = (_Float16)0.f;
    long Kf = (long)Cin * KH * KW;
    if (m < Cout && k < Kf) {
        int kw = (int)(k % KW);
        int kh = (int)((k / KW) % KH);
        int ci = (int)(k / ((long)KW * KH));
        v = (_Float16)w[(((long)ci * Cout + m) * KH + (KH - 1 - kh)) * KW + (KW - 1 - kw)];
    }
    dst[i] = v;
}

// im2col producing the TRANSPOSED patch matrix BT [Npc x Kp] (N-major, K inner)
__global__ void im2colT_k(_Float16* __restrict__ dst, const float* __restrict__ src,
                          int Cin, int H, int W, int KH, int stride, int pad,
                          int Ho, int Wo, long cstride, long bstride,
                          int b0, long Nc, long Kp, long Npc)
{
    long i = blockIdx.x * (long)blockDim.x + threadIdx.x;
    if (i >= Npc * Kp) return;
    long col = i / Kp;   // N index (output pixel)
    long row = i % Kp;   // K index (cin,kh,kw)
    _Float16 v = (_Float16)0.f;
    long Kf = (long)Cin * KH * KH;
    if (row < Kf && col < Nc) {
        long HoWo = (long)Ho * Wo;
        int kw = (int)(row % KH);
        int kh = (int)((row / KH) % KH);
        int ci = (int)(row / ((long)KH * KH));
        int wo = (int)(col % Wo);
        int ho = (int)((col / Wo) % Ho);
        int b  = b0 + (int)(col / HoWo);
        int hi = ho * stride - pad + kh;
        int wi = wo * stride - pad + kw;
        if (hi >= 0 && hi < H && wi >= 0 && wi < W)
            v = (_Float16)src[(long)ci * cstride + (long)b * bstride + (long)hi * W + wi];
    }
    dst[i] = v;
}

// ---------------------------------------------------------------------------
// Elementwise / epilogue kernels
// ---------------------------------------------------------------------------
__global__ void bias_act_k(float* x, const float* bias, int C, long ld, int mode)
{
    long i = blockIdx.x * (long)blockDim.x + threadIdx.x;
    if (i >= (long)C * ld) return;
    int c = (int)(i / ld);
    float v = x[i] + (bias ? bias[c] : 0.f);
    if (mode == 1)      v = fmaxf(v, 0.f);
    else if (mode == 2) v = v > 0.f ? v : 0.2f * v;
    else if (mode == 3) v = tanhf(v);
    x[i] = v;
}

__global__ void bn_stats_k(const float* x, long n, float* mean, float* var)
{
    int c = blockIdx.x;
    const float* p = x + (long)c * n;
    float s = 0.f, s2 = 0.f;
    for (long i = threadIdx.x; i < n; i += blockDim.x) { float v = p[i]; s += v; s2 += v * v; }
    __shared__ float sh[TPB], sh2[TPB];
    sh[threadIdx.x] = s; sh2[threadIdx.x] = s2;
    __syncthreads();
    for (int o = TPB / 2; o > 0; o >>= 1) {
        if ((int)threadIdx.x < o) { sh[threadIdx.x] += sh[threadIdx.x + o]; sh2[threadIdx.x] += sh2[threadIdx.x + o]; }
        __syncthreads();
    }
    if (threadIdx.x == 0) {
        float m = sh[0] / (float)n;
        mean[c] = m;
        var[c]  = sh2[0] / (float)n - m * m;
    }
}

__global__ void bn_apply_k(float* x, const float* mean, const float* var,
                           const float* g, const float* b, int C, long n)
{
    long i = blockIdx.x * (long)blockDim.x + threadIdx.x;
    if (i >= (long)C * n) return;
    int c = (int)(i / n);
    float v = (x[i] - mean[c]) * rsqrtf(var[c] + 1e-5f) * g[c] + b[c];
    x[i] = v > 0.f ? v : 0.2f * v;   // batchnorm is always followed by lrelu here
}

__global__ void ew_unary_k(float* x, long n, int mode)
{
    long i = blockIdx.x * (long)blockDim.x + threadIdx.x;
    if (i >= n) return;
    float v = x[i];
    if (mode == 1)      v = fmaxf(v, 0.f);
    else if (mode == 2) v = v > 0.f ? v : 0.2f * v;
    else if (mode == 3) v = tanhf(v);
    x[i] = v;
}

__global__ void ew_unary_out_k(float* dst, const float* src, long n, int mode)
{
    long i = blockIdx.x * (long)blockDim.x + threadIdx.x;
    if (i >= n) return;
    float v = src[i];
    if (mode == 1)      v = fmaxf(v, 0.f);
    else if (mode == 2) v = v > 0.f ? v : 0.2f * v;
    dst[i] = v;
}

__global__ void ew_add_k(float* x, const float* y, long n)
{
    long i = blockIdx.x * (long)blockDim.x + threadIdx.x;
    if (i < n) x[i] += y[i];
}

__global__ void gamma_res_k(float* o, const float* x, const float* gamma, long n)
{
    long i = blockIdx.x * (long)blockDim.x + threadIdx.x;
    if (i < n) o[i] = gamma[0] * o[i] + x[i];
}

__global__ void softmax_row_k(float* att, int n, long ld)
{
    float* p = att + (long)blockIdx.x * ld;
    __shared__ float red[TPB];
    float m = -3.0e38f;
    for (int j = threadIdx.x; j < n; j += TPB) m = fmaxf(m, p[j]);
    red[threadIdx.x] = m; __syncthreads();
    for (int o = TPB / 2; o > 0; o >>= 1) {
        if ((int)threadIdx.x < o) red[threadIdx.x] = fmaxf(red[threadIdx.x], red[threadIdx.x + o]);
        __syncthreads();
    }
    m = red[0]; __syncthreads();
    float s = 0.f;
    for (int j = threadIdx.x; j < n; j += TPB) { float e = __expf(p[j] - m); p[j] = e; s += e; }
    red[threadIdx.x] = s; __syncthreads();
    for (int o = TPB / 2; o > 0; o >>= 1) {
        if ((int)threadIdx.x < o) red[threadIdx.x] += red[threadIdx.x + o];
        __syncthreads();
    }
    float inv = 1.f / red[0];
    for (int j = threadIdx.x; j < n; j += TPB) p[j] *= inv;
}

// zero-dilate by 2 for conv-transpose: dst [C][B*(2H-1)*(2W-1)]
__global__ void upsample2x_k(float* dst, const float* src, int C, int Bn,
                             int H, int W, long ldsrc)
{
    int Hu = 2 * H - 1, Wu = 2 * W - 1;
    long tot = (long)C * Bn * Hu * Wu;
    long i = blockIdx.x * (long)blockDim.x + threadIdx.x;
    if (i >= tot) return;
    int wu = (int)(i % Wu);
    long t = i / Wu;
    int hu = (int)(t % Hu); t /= Hu;
    int b  = (int)(t % Bn);
    int c  = (int)(t / Bn);
    float v = 0.f;
    if (!(hu & 1) && !(wu & 1))
        v = src[(long)c * ldsrc + (long)b * H * W + (long)(hu >> 1) * W + (wu >> 1)];
    dst[i] = v;
}

// VQ: per spatial column, argmin over 512 codes of dim 64, gather, stats
__global__ void vq_assign_k(const float* zc, long N, const float* emb,
                            float* qb, float* probs, float* acc)
{
    long i = blockIdx.x * (long)blockDim.x + threadIdx.x;
    if (i >= N) return;
    float z[64];
#pragma unroll
    for (int d = 0; d < 64; ++d) z[d] = zc[(long)d * N + i];
    int best = 0; float bd = 3.4e38f;
    for (int k = 0; k < 512; ++k) {
        const float* e = emb + (long)k * 64;
        float dist = 0.f;
#pragma unroll 8
        for (int d = 0; d < 64; ++d) { float df = z[d] - e[d]; dist += df * df; }
        if (dist < bd) { bd = dist; best = k; }
    }
    const float* e = emb + (long)best * 64;
    for (int d = 0; d < 64; ++d) qb[(long)d * N + i] = e[d];
    atomicAdd(probs + best, 1.f);
    atomicAdd(acc, bd);
}

__global__ void add_loss_k(float* lossAcc, const float* acc, float scale)
{
    *lossAcc += scale * (*acc);
}

__global__ void vq_perp_k(const float* probs, float Ninv, float* outp)
{
    __shared__ float red[512];
    int k = threadIdx.x;
    float p = probs[k] * Ninv;
    red[k] = p * __logf(p + 1e-10f);
    __syncthreads();
    for (int o = 256; o > 0; o >>= 1) {
        if (k < o) red[k] += red[k + o];
        __syncthreads();
    }
    if (k == 0) *outp = __expf(-red[0]);
}

__global__ void write_loss_k(float* d, const float* a) { d[0] = a[0]; }

__global__ void to_nchw_k(float* dst, const float* src, int Bn, int C, int HW)
{
    long tot = (long)Bn * C * HW;
    long i = blockIdx.x * (long)blockDim.x + threadIdx.x;
    if (i >= tot) return;
    int hw = (int)(i % HW);
    long t = i / HW;
    int c = (int)(t % C);
    int b = (int)(t / C);
    dst[i] = src[(long)c * Bn * HW + (long)b * HW + hw];
}

// ---------------------------------------------------------------------------
// Host-side orchestration
// ---------------------------------------------------------------------------
struct Ctx {
    hipStream_t s;
    char* base;
    size_t off;
    void* al(size_t bytes) { size_t a = (off + 255) & ~(size_t)255; off = a + bytes; return base + a; }
    float*    f(size_t n)   { return (float*)al(n * sizeof(float)); }
    _Float16* h16(size_t n) { return (_Float16*)al(n * sizeof(_Float16)); }
};

struct ConvP { const float* w; const float* b; };
struct ResL  { ConvP c1, c2; };
struct VqP   { ConvP pre; const float* emb; ConvP post; };

// Activations live as [C][B*H*W] (channel-major, then batch, then spatial).
static float* conv_layer(Ctx& c, const float* x, long cstride, long bstride,
                         int Cin, int Bn, int H, int W,
                         const float* w, const float* bias, int Cout,
                         int KH, int stride, int pad, int actmode, bool tw,
                         int* oH, int* oW)
{
    const int Ho = (H + 2 * pad - KH) / stride + 1;
    const int Wo = (W + 2 * pad - KH) / stride + 1;
    const long HoWo = (long)Ho * Wo;
    const long Ncols = (long)Bn * HoWo;
    const long Kf = (long)Cin * KH * KH;
    const long Kp = rnd(Kf, 32), Mp = rnd(Cout, 64);

    float* out = c.f((size_t)Cout * Ncols);
    size_t mark = c.off;

    _Float16* Ap = c.h16((size_t)Mp * Kp);
    if (!tw) pack_f16_k<<<GRID1(Mp * Kp), TPB, 0, c.s>>>(Ap, w, Cout, Kf, Kf, 1, Mp, Kp);
    else     pack_wT_k<<<GRID1(Mp * Kp), TPB, 0, c.s>>>(Ap, w, Cin, Cout, KH, KH, Kp, Mp);

    int bchunk = Bn;
    while ((long)bchunk * HoWo > 65536 && bchunk > 1) bchunk >>= 1;
    for (int b0 = 0; b0 < Bn; b0 += bchunk) {
        int bc = (Bn - b0 < bchunk) ? (Bn - b0) : bchunk;
        long Nc = (long)bc * HoWo;
        long Npc = rnd(Nc, 128);
        size_t m2 = c.off;
        _Float16* Bp = c.h16((size_t)Kp * Npc);
        im2colT_k<<<GRID1(Npc * Kp), TPB, 0, c.s>>>(Bp, x, Cin, H, W, KH, stride, pad,
                                                    Ho, Wo, cstride, bstride, b0, Nc, Kp, Npc);
        dim3 grid((unsigned)(Npc / 128), (unsigned)(Mp / 64));
        gemm_wmma_f16<<<grid, 256, 0, c.s>>>(Ap, Bp, out + (long)b0 * HoWo,
                                             Cout, (int)Nc, (int)Kp, (int)Kp, (int)Kp, Ncols);
        c.off = m2;
    }
    c.off = mark;

    bias_act_k<<<GRID1((long)Cout * Ncols), TPB, 0, c.s>>>(out, bias, Cout, Ncols, actmode);
    *oH = Ho; *oW = Wo;
    return out;
}

static void bn_lrelu(Ctx& c, float* x, int C, long n, const float* g, const float* b)
{
    size_t mark = c.off;
    float* mv = c.f((size_t)2 * C);
    bn_stats_k<<<C, TPB, 0, c.s>>>(x, n, mv, mv + C);
    bn_apply_k<<<GRID1((long)C * n), TPB, 0, c.s>>>(x, mv, mv + C, g, b, C, n);
    c.off = mark;
}

static float* concat2(Ctx& c, const float* a, int Ca, const float* b, int Cb, long N)
{
    float* o = c.f((size_t)(Ca + Cb) * N);
    (void)hipMemcpyAsync(o, a, (size_t)Ca * N * sizeof(float), hipMemcpyDeviceToDevice, c.s);
    (void)hipMemcpyAsync(o + (size_t)Ca * N, b, (size_t)Cb * N * sizeof(float), hipMemcpyDeviceToDevice, c.s);
    return o;
}

static float* attn_layer(Ctx& c, const float* x, int C, int Bn, int n,
                         const ConvP& pq, const ConvP& pk, const ConvP& pv,
                         const float* gamma, int Cq)
{
    const long N = (long)Bn * n;
    int oh, ow;
    float* q = conv_layer(c, x, N, n, C, Bn, n, 1, pq.w, pq.b, Cq, 1, 1, 0, 0, false, &oh, &ow);
    float* k = conv_layer(c, x, N, n, C, Bn, n, 1, pk.w, pk.b, Cq, 1, 1, 0, 0, false, &oh, &ow);
    float* v = conv_layer(c, x, N, n, C, Bn, n, 1, pv.w, pv.b, C,  1, 1, 0, 0, false, &oh, &ow);

    float* out = c.f((size_t)C * N);
    size_t mark = c.off;
    const long np  = rnd(n, 128);
    const long Cqp = rnd(Cq, 32);
    const long Cp  = rnd(C, 64);
    float*    att = c.f((size_t)n * np);
    _Float16* qT  = c.h16((size_t)np * Cqp);   // A:  q^T slice [n x Cq]
    _Float16* kT  = c.h16((size_t)np * Cqp);   // BT: k slice transposed = [n x Cq]
    _Float16* aB  = c.h16((size_t)np * np);    // BT for gemm2 = attn itself [n x n]
    _Float16* vS  = c.h16((size_t)Cp * np);    // A:  V slice [C x n]

    for (int b = 0; b < Bn; ++b) {
        // attn = (q^T)(k): A=[n x Cq], BT=[n x Cq] (both K-major)
        pack_f16_k<<<GRID1(np * Cqp), TPB, 0, c.s>>>(qT, q + (long)b * n, n, Cq, 1, N, np, Cqp);
        pack_f16_k<<<GRID1(np * Cqp), TPB, 0, c.s>>>(kT, k + (long)b * n, n, Cq, 1, N, np, Cqp);
        dim3 g1((unsigned)(np / 128), (unsigned)(np / 64));
        gemm_wmma_f16<<<g1, 256, 0, c.s>>>(qT, kT, att, n, n, (int)Cqp, (int)Cqp, (int)Cqp, np);
        softmax_row_k<<<n, TPB, 0, c.s>>>(att, n, np);
        // out = V * attn^T: A=[C x n], BT=(attn^T)^T = attn [n x n]
        pack_f16_k<<<GRID1(np * np), TPB, 0, c.s>>>(aB, att, n, n, np, 1, np, np);
        pack_f16_k<<<GRID1(Cp * np), TPB, 0, c.s>>>(vS, v + (long)b * n, C, n, N, 1, Cp, np);
        dim3 g2((unsigned)(np / 128), (unsigned)(Cp / 64));
        gemm_wmma_f16<<<g2, 256, 0, c.s>>>(vS, aB, out + (long)b * n, C, n, (int)np, (int)np, (int)np, N);
    }
    gamma_res_k<<<GRID1((long)C * N), TPB, 0, c.s>>>(out, x, gamma, (long)C * N);
    c.off = mark;
    return out;
}

static float* res_stack_f(Ctx& c, const float* x0, int C, int Bn, int H, int W,
                          const ResL* L, int nl)
{
    const long N = (long)Bn * H * W;
    float* cur = c.f((size_t)C * N);
    (void)hipMemcpyAsync(cur, x0, (size_t)C * N * sizeof(float), hipMemcpyDeviceToDevice, c.s);
    int oh, ow;
    for (int l = 0; l < nl; ++l) {
        size_t mark = c.off;
        float* t = c.f((size_t)C * N);
        ew_unary_out_k<<<GRID1((long)C * N), TPB, 0, c.s>>>(t, cur, (long)C * N, 1);
        // conv(relu(x)) 3x3 -> relu fused on output (it only feeds the next conv)
        float* h1 = conv_layer(c, t, N, (long)H * W, C, Bn, H, W, L[l].c1.w, L[l].c1.b,
                               32, 3, 1, 1, 1, false, &oh, &ow);
        float* h2 = conv_layer(c, h1, N, (long)H * W, 32, Bn, H, W, L[l].c2.w, L[l].c2.b,
                               C, 1, 1, 0, 0, false, &oh, &ow);
        ew_add_k<<<GRID1((long)C * N), TPB, 0, c.s>>>(cur, h2, (long)C * N);
        c.off = mark;
    }
    ew_unary_k<<<GRID1((long)C * N), TPB, 0, c.s>>>(cur, (long)C * N, 1);
    return cur;
}

// ConvTranspose2d(k=4,s=2,p=1) == conv(k=4,s=1,p=2) over zero-dilated input
static float* convT_layer(Ctx& c, const float* x, int Cin, int Bn, int H, int W,
                          const float* w, const float* bias, int Cout, int act,
                          int* oH, int* oW)
{
    const int Hu = 2 * H - 1, Wu = 2 * W - 1;
    float* up = c.f((size_t)Cin * Bn * Hu * Wu);
    upsample2x_k<<<GRID1((long)Cin * Bn * Hu * Wu), TPB, 0, c.s>>>(up, x, Cin, Bn, H, W, (long)Bn * H * W);
    return conv_layer(c, up, (long)Bn * Hu * Wu, (long)Hu * Wu, Cin, Bn, Hu, Wu,
                      w, bias, Cout, 4, 1, 2, act, true, oH, oW);
}

static float* vq_layer(Ctx& c, const float* x, int Bn, int n, const VqP& p,
                       float* lossAcc, float* perpOut)
{
    const long N = (long)Bn * n;
    int oh, ow;
    float* zc = conv_layer(c, x, N, n, 128, Bn, n, 1, p.pre.w, p.pre.b, 64, 1, 1, 0, 0, false, &oh, &ow);
    float* qb = c.f((size_t)64 * N);
    size_t mark = c.off;
    float* probs = c.f(512);
    float* acc   = c.f(1);
    (void)hipMemsetAsync(probs, 0, 512 * sizeof(float), c.s);
    (void)hipMemsetAsync(acc,   0, sizeof(float), c.s);
    vq_assign_k<<<GRID1(N), TPB, 0, c.s>>>(zc, N, p.emb, qb, probs, acc);
    // forward loss = (1 + 0.25) * mean((q - zc)^2)
    add_loss_k<<<1, 1, 0, c.s>>>(lossAcc, acc, 1.25f / (float)(N * 64));
    vq_perp_k<<<1, 512, 0, c.s>>>(probs, 1.f / (float)N, perpOut);
    c.off = mark;
    // forward value of straight-through q_st == q
    return conv_layer(c, qb, N, n, 64, Bn, n, 1, p.post.w, p.post.b, 128, 1, 1, 0, 0, false, &oh, &ow);
}

extern "C" void kernel_launch(void* const* d_in, const int* in_sizes, int n_in,
                              void* d_out, int out_size, void* d_ws, size_t ws_size,
                              hipStream_t stream)
{
    (void)in_sizes; (void)n_in; (void)out_size; (void)ws_size;
    Ctx c{stream, (char*)d_ws, 0};
    float* dOut = (float*)d_out;
    float* lossAcc = c.f(1);
    (void)hipMemsetAsync(lossAcc, 0, sizeof(float), stream);

    int cur = 0;
    auto nxt = [&]() { return (const float*)d_in[cur++]; };
    auto rdC = [&]() { ConvP p; p.w = nxt(); p.b = nxt(); return p; };
    struct Bn2 { const float* g; const float* b; };
    auto rdB = [&]() { Bn2 p; p.g = nxt(); p.b = nxt(); return p; };
    struct Sa { ConvP q, k, v; const float* gamma; };
    auto rdS = [&]() { Sa s; s.q = rdC(); s.k = rdC(); s.v = rdC(); s.gamma = nxt(); return s; };
    auto rdR = [&](ResL* r) { for (int i = 0; i < 2; ++i) { r[i].c1 = rdC(); r[i].c2 = rdC(); } };
    auto rdV = [&]() { VqP v; v.pre = rdC(); v.emb = nxt(); v.post = rdC(); return v; };

    const float* images = nxt();
    ConvP c1c = rdC(); Bn2 c1b = rdB();
    ConvP s2c = rdC(); Bn2 s2b = rdB(); Sa s2s = rdS();
    ConvP s3c = rdC(); Bn2 s3b = rdB(); Sa s3s = rdS();
    ResL saResEnc[2]; rdR(saResEnc);
    VqP vqAtt = rdV();
    ConvP n2c = rdC(); Bn2 n2b = rdB();
    ConvP n3c = rdC(); Bn2 n3b = rdB();
    ResL resEnc[2]; rdR(resEnc);
    VqP vqNorm = rdV();
    ResL saResDec[2]; rdR(saResDec);
    Sa attD1 = rdS();
    ConvP saDc1 = rdC(); Sa saDc1s = rdS();
    ConvP saDc2 = rdC();
    ResL resDec[2]; rdR(resDec);
    ConvP dc1 = rdC(); ConvP dc2 = rdC(); ConvP dcR = rdC();

    const int B = 32;
    int oh, ow;

    // encoder
    float* g = conv_layer(c, images, 128L * 128, 3L * 128 * 128, 3, B, 128, 128,
                          c1c.w, c1c.b, 32, 4, 2, 1, 0, false, &oh, &ow);       // [32][B*64*64]
    long Ng = (long)B * 64 * 64;
    bn_lrelu(c, g, 32, Ng, c1b.g, c1b.b);

    float* s1 = conv_layer(c, g, Ng, 64L * 64, 32, B, 64, 64, s2c.w, s2c.b, 64, 4, 2, 1, 0, false, &oh, &ow);
    long N1 = (long)B * 1024;
    bn_lrelu(c, s1, 64, N1, s2b.g, s2b.b);
    float* s1a = attn_layer(c, s1, 64, B, 1024, s2s.q, s2s.k, s2s.v, s2s.gamma, 32);
    ew_unary_k<<<GRID1(64 * N1), TPB, 0, stream>>>(s1a, 64 * N1, 2);

    float* s2 = conv_layer(c, s1a, N1, 1024, 64, B, 32, 32, s3c.w, s3c.b, 128, 4, 2, 1, 0, false, &oh, &ow);
    long N2 = (long)B * 256;
    bn_lrelu(c, s2, 128, N2, s3b.g, s3b.b);
    float* s2a = attn_layer(c, s2, 128, B, 256, s3s.q, s3s.k, s3s.v, s3s.gamma, 64);
    ew_unary_k<<<GRID1(128 * N2), TPB, 0, stream>>>(s2a, 128 * N2, 2);
    float* saRes = res_stack_f(c, s2a, 128, B, 16, 16, saResEnc, 2);

    float* n1 = conv_layer(c, g, Ng, 64L * 64, 32, B, 64, 64, n2c.w, n2c.b, 64, 4, 2, 1, 0, false, &oh, &ow);
    bn_lrelu(c, n1, 64, N1, n2b.g, n2b.b);
    float* cat1 = concat2(c, n1, 64, s1a, 64, N1);
    float* n2a = conv_layer(c, cat1, N1, 1024, 128, B, 32, 32, n3c.w, n3c.b, 128, 4, 2, 1, 0, false, &oh, &ow);
    bn_lrelu(c, n2a, 128, N2, n3b.g, n3b.b);
    float* nRes = res_stack_f(c, n2a, 128, B, 16, 16, resEnc, 2);

    // vector quantization (perp values go straight to their output slots)
    float* nEnc = vq_layer(c, nRes, B, 256, vqNorm, lossAcc, dOut + 1 + 1572864);
    float* sEnc = vq_layer(c, saRes, B, 256, vqAtt, lossAcc, dOut + 1 + 1572864 + 1);

    // decoder
    float* saRD = res_stack_f(c, sEnc, 128, B, 16, 16, saResDec, 2);
    float* att1 = attn_layer(c, saRD, 128, B, 256, attD1.q, attD1.k, attD1.v, attD1.gamma, 64);
    float* t1 = convT_layer(c, att1, 128, B, 16, 16, saDc1.w, saDc1.b, 64, 2, &oh, &ow);   // 32x32
    float* saD1 = attn_layer(c, t1, 64, B, 1024, saDc1s.q, saDc1s.k, saDc1s.v, saDc1s.gamma, 16);
    ew_unary_k<<<GRID1(64 * N1), TPB, 0, stream>>>(saD1, 64 * N1, 2);

    float* nRD = res_stack_f(c, nEnc, 128, B, 16, 16, resDec, 2);
    float* cat2 = concat2(c, nRD, 128, att1, 128, N2);
    float* nD1 = convT_layer(c, cat2, 256, B, 16, 16, dc1.w, dc1.b, 64, 2, &oh, &ow);      // 32x32
    float* nD2 = convT_layer(c, nD1, 64, B, 32, 32, dc2.w, dc2.b, 32, 2, &oh, &ow);        // 64x64
    float* cat3 = concat2(c, nD1, 64, saD1, 64, N1);
    float* saD2 = convT_layer(c, cat3, 128, B, 32, 32, saDc2.w, saDc2.b, 32, 2, &oh, &ow); // 64x64
    float* cat4 = concat2(c, saD2, 32, nD2, 32, Ng);
    float* rec = convT_layer(c, cat4, 64, B, 64, 64, dcR.w, dcR.b, 3, 3, &oh, &ow);        // 128x128

    // outputs: [loss, recon(NCHW), n_perp, s_perp]
    write_loss_k<<<1, 1, 0, stream>>>(dOut, lossAcc);
    to_nchw_k<<<GRID1((long)B * 3 * 16384), TPB, 0, stream>>>(dOut + 1, rec, B, 3, 16384);
}
